// Interaction_Net_49400713838809
// MI455X (gfx1250) — matrix operand order repaired and checked by reading, not compiled, not run on hardware
//
#include <hip/hip_runtime.h>
#include <math.h>

typedef __bf16 bf16_t;
typedef __attribute__((ext_vector_type(16))) __bf16 v16bf;
typedef __attribute__((ext_vector_type(8)))  __bf16 v8bf;
typedef __attribute__((ext_vector_type(8)))  float  v8f;
typedef unsigned int v4u __attribute__((ext_vector_type(4)));
typedef int v8i __attribute__((ext_vector_type(8)));
typedef int v4i __attribute__((ext_vector_type(4)));

#define DEV __device__ __forceinline__

// ---------------------------------------------------------------------------
// WMMA tile helpers (CDNA5 §7.12.2 layouts, wave32)
// ---------------------------------------------------------------------------

// A: 16x32 bf16 tile from fp32 row-major X[M,ldx], rows m0..m0+15, K kb..kb+31.
DEV v16bf load_A_f32(const float* __restrict__ X, int ldx, int m0, int kb, int lane) {
  const int m = m0 + (lane & 15);
  const int khalf = (lane >> 4) << 3;          // 0 or 8
  const float4* p = (const float4*)(X + (size_t)m * ldx + kb + khalf);
  float4 f0 = p[0], f1 = p[1];                  // K = kb+khalf+0..7
  float4 f2 = p[4], f3 = p[5];                  // K = kb+khalf+16..23
  v16bf a;
  a[0]=(__bf16)f0.x; a[1]=(__bf16)f0.y; a[2]=(__bf16)f0.z; a[3]=(__bf16)f0.w;
  a[4]=(__bf16)f1.x; a[5]=(__bf16)f1.y; a[6]=(__bf16)f1.z; a[7]=(__bf16)f1.w;
  a[8]=(__bf16)f2.x; a[9]=(__bf16)f2.y; a[10]=(__bf16)f2.z; a[11]=(__bf16)f2.w;
  a[12]=(__bf16)f3.x; a[13]=(__bf16)f3.y; a[14]=(__bf16)f3.z; a[15]=(__bf16)f3.w;
  return a;
}

// B tile (32x16) from an LDS-resident weight block stored row-major [rows][ldw] bf16.
DEV v16bf load_B_lds(const bf16_t* wlds, int ldw, int n0, int kb, int lane) {
  const int n = n0 + (lane & 15);
  const int k0 = kb + ((lane >> 4) << 3);       // 0 or 8
  const bf16_t* p = wlds + (size_t)n * ldw + k0;
  v8bf b0 = *(const v8bf*)p;
  v8bf b1 = *(const v8bf*)(p + 16);
  v16bf b;
#pragma unroll
  for (int e = 0; e < 8; ++e) { b[e] = b0[e]; b[8 + e] = b1[e]; }
  return b;
}

// B tile straight from global bf16 weight [Nout, ldw]
DEV v16bf load_B(const bf16_t* __restrict__ Wb, int ldw, int n0, int kb, int lane) {
  const int n = n0 + (lane & 15);
  const int k0 = kb + ((lane >> 4) << 3);
  const bf16_t* p = Wb + (size_t)n * ldw + k0;
  v8bf b0 = *(const v8bf*)p;
  v8bf b1 = *(const v8bf*)(p + 16);
  v16bf b;
#pragma unroll
  for (int e = 0; e < 8; ++e) { b[e] = b0[e]; b[8 + e] = b1[e]; }
  return b;
}

DEV v8f wmma_bf16(v16bf a, v16bf b, v8f c) {
  return __builtin_amdgcn_wmma_f32_16x16x32_bf16(false, a, false, b, (short)0, c, false, false);
}

// ---------------------------------------------------------------------------
// TDM: async-DMA a 128x128 bf16 weight block (32KB) into LDS.
// D# built per CDNA5 ISA: 1-D tensor of 16384 2-byte
// elements; group0 = {count=1, lds_addr, global_addr[56:0], type=2};
// group1 = {data_size=1(2B), tensor_dim0=16384, tile_dim0=16384}.
// Issued by one wave; completion via TENSORcnt, published with a barrier.
// ---------------------------------------------------------------------------
DEV void tdm_stage_w128(const bf16_t* __restrict__ W, bf16_t* wlds_sh) {
  const unsigned int lds_addr = (unsigned int)(size_t)(void*)wlds_sh;  // low 32b = LDS offset
  const unsigned long long ga = (unsigned long long)(size_t)W;
  v4u g0;
  g0[0] = 1u;                                            // count=1 (valid user D#)
  g0[1] = lds_addr;                                      // lds_addr[31:0]
  g0[2] = (unsigned int)(ga & 0xffffffffu);              // global_addr[31:0]
  g0[3] = (unsigned int)((ga >> 32) & 0x1ffffffu) | (2u << 30);  // addr[56:32] | type=2
  v8i g1;
  g1[0] = 0x00010000;          // workgroup_mask=0, data_size=1 (2 bytes)
  g1[1] = 0;                   // no atomic barrier
  g1[2] = (int)(16384u << 16); // tensor_dim0[15:0]=16384 (bits 79:64... [63:48] span)
  g1[3] = (int)(16384u << 16); // tensor_dim1[15:0]=0 here unused; tile_dim0=16384 in [127:112]
  g1[2] |= 0;                  // tensor_dim0 high bits zero
  g1[4] = 0;                   // tile_dim1=0 (unused), tile_dim2=0
  g1[5] = 16384;               // tensor_dim0_stride[31:0]
  g1[6] = 0;
  g1[7] = 0;
  v4i g2 = {0, 0, 0, 0};
  v4i g3 = {0, 0, 0, 0};
#if defined(__clang_major__) && __clang_major__ >= 23
  v8i g4 = {0, 0, 0, 0, 0, 0, 0, 0};
  __builtin_amdgcn_tensor_load_to_lds(g0, g1, g2, g3, g4, 0);
#else
  __builtin_amdgcn_tensor_load_to_lds(g0, g1, g2, g3, 0);
#endif
}

// ---------------------------------------------------------------------------
// GEMM (K = Nout = 128), weight TDM-staged in LDS, one 16x128 tile per wave.
// op 0: Y = X@W^T + b
// op 2: Y = X@W^T + b + aux                (residual)
// op 3: Y = X * sigmoid(X@W^T + b + aux)   (GRU-style gate)
// ---------------------------------------------------------------------------
__global__ void __launch_bounds__(128) gemm128_kernel(
    const float* __restrict__ X, const bf16_t* __restrict__ Wb,
    const float* __restrict__ bias, const float* __restrict__ aux,
    float* __restrict__ Y, int M, int op) {
  __shared__ bf16_t wlds[128 * 128];
  if (threadIdx.x < 32) {                       // wave 0 issues the DMA
    tdm_stage_w128(Wb, wlds);
    __builtin_amdgcn_s_wait_tensorcnt(0);
  }
  __syncthreads();

  const int lane = threadIdx.x & 31;
  const int wave = threadIdx.x >> 5;
  const int m0 = (blockIdx.x * 4 + wave) * 16;
  if (m0 >= M) return;

  v8f zero = {};
  v8f acc[8];
#pragma unroll
  for (int t = 0; t < 8; ++t) acc[t] = zero;

#pragma unroll
  for (int kb = 0; kb < 128; kb += 32) {
    v16bf a = load_A_f32(X, 128, m0, kb, lane);
#pragma unroll
    for (int g = 0; g < 2; ++g) {
      v16bf bt[4];
#pragma unroll
      for (int u = 0; u < 4; ++u) bt[u] = load_B_lds(wlds, 128, (g * 4 + u) * 16, kb, lane);
#pragma unroll
      for (int u = 0; u < 4; ++u) acc[g * 4 + u] = wmma_bf16(a, bt[u], acc[g * 4 + u]);
    }
  }

  const int rbase = m0 + ((lane >> 4) << 3);
  const int coll = lane & 15;
#pragma unroll
  for (int t = 0; t < 8; ++t) {
    const int col = t * 16 + coll;
    const float bc = bias[col];
#pragma unroll
    for (int r = 0; r < 8; ++r) {
      const size_t idx = (size_t)(rbase + r) * 128 + col;
      float v = acc[t][r] + bc;
      if (op == 2) {
        v += aux[idx];
      } else if (op == 3) {
        float g = 1.f / (1.f + __expf(-(v + aux[idx])));
        v = X[idx] * g;
      }
      Y[idx] = v;
    }
  }
}

// ---------------------------------------------------------------------------
// Dual GEMM: one pass over X produces Y1 = X@W1^T+b1 and Y2 = X@W2^T+b2.
// A tiles (all 4 K-blocks) kept in registers; W1 then W2 TDM-staged in LDS.
// ---------------------------------------------------------------------------
__global__ void __launch_bounds__(128) gemm128_dual_kernel(
    const float* __restrict__ X,
    const bf16_t* __restrict__ W1b, const float* __restrict__ b1v, float* __restrict__ Y1,
    const bf16_t* __restrict__ W2b, const float* __restrict__ b2v, float* __restrict__ Y2,
    int M) {
  __shared__ bf16_t wlds[128 * 128];
  const int lane = threadIdx.x & 31;
  const int wave = threadIdx.x >> 5;
  const int m0 = (blockIdx.x * 4 + wave) * 16;
  const bool valid = (m0 < M);

  if (threadIdx.x < 32) {
    tdm_stage_w128(W1b, wlds);
    __builtin_amdgcn_s_wait_tensorcnt(0);
  }

  v16bf a[4];
  if (valid) {
#pragma unroll
    for (int k4 = 0; k4 < 4; ++k4) a[k4] = load_A_f32(X, 128, m0, k4 * 32, lane);
  }
  __syncthreads();

  const int rbase = m0 + ((lane >> 4) << 3);
  const int coll = lane & 15;
  v8f zero = {};
  v8f acc[8];

  if (valid) {
#pragma unroll
    for (int t = 0; t < 8; ++t) acc[t] = zero;
#pragma unroll
    for (int k4 = 0; k4 < 4; ++k4)
#pragma unroll
      for (int g = 0; g < 2; ++g) {
        v16bf bt[4];
#pragma unroll
        for (int u = 0; u < 4; ++u) bt[u] = load_B_lds(wlds, 128, (g * 4 + u) * 16, k4 * 32, lane);
#pragma unroll
        for (int u = 0; u < 4; ++u) acc[g * 4 + u] = wmma_bf16(a[k4], bt[u], acc[g * 4 + u]);
      }
#pragma unroll
    for (int t = 0; t < 8; ++t) {
      const int col = t * 16 + coll;
      const float bc = b1v[col];
#pragma unroll
      for (int r = 0; r < 8; ++r)
        Y1[(size_t)(rbase + r) * 128 + col] = acc[t][r] + bc;
    }
  }
  __syncthreads();                 // everyone done reading W1
  if (threadIdx.x < 32) {
    tdm_stage_w128(W2b, wlds);
    __builtin_amdgcn_s_wait_tensorcnt(0);
  }
  __syncthreads();

  if (valid) {
#pragma unroll
    for (int t = 0; t < 8; ++t) acc[t] = zero;
#pragma unroll
    for (int k4 = 0; k4 < 4; ++k4)
#pragma unroll
      for (int g = 0; g < 2; ++g) {
        v16bf bt[4];
#pragma unroll
        for (int u = 0; u < 4; ++u) bt[u] = load_B_lds(wlds, 128, (g * 4 + u) * 16, k4 * 32, lane);
#pragma unroll
        for (int u = 0; u < 4; ++u) acc[g * 4 + u] = wmma_bf16(a[k4], bt[u], acc[g * 4 + u]);
      }
#pragma unroll
    for (int t = 0; t < 8; ++t) {
      const int col = t * 16 + coll;
      const float bc = b2v[col];
#pragma unroll
      for (int r = 0; r < 8; ++r)
        Y2[(size_t)(rbase + r) * 128 + col] = acc[t][r] + bc;
    }
  }
}

// ---------------------------------------------------------------------------
// Fused FFN: out = XT + relu(HN @ Wm1^T + bm1) @ Wm2^T + bm2
// 2 waves/block, 16x512 intermediate per wave staged in LDS as bf16.
// ---------------------------------------------------------------------------
__global__ void __launch_bounds__(64) ffn_fused_kernel(
    const float* __restrict__ HN, const float* __restrict__ XT,
    const bf16_t* __restrict__ Wm1b, const float* __restrict__ bm1,
    const bf16_t* __restrict__ Wm2b, const float* __restrict__ bm2,
    float* __restrict__ out, int M) {
  __shared__ bf16_t lds[2 * 16 * 512];
  const int lane = threadIdx.x & 31;
  const int wave = threadIdx.x >> 5;
  const int m0 = (blockIdx.x * 2 + wave) * 16;
  bf16_t* ldsw = lds + wave * 16 * 512;
  v8f zero = {};

  if (m0 < M) {
    for (int nchunk = 0; nchunk < 4; ++nchunk) {
      v8f acc[8];
#pragma unroll
      for (int t = 0; t < 8; ++t) acc[t] = zero;
      for (int kb = 0; kb < 128; kb += 32) {
        v16bf a = load_A_f32(HN, 128, m0, kb, lane);
#pragma unroll
        for (int g = 0; g < 2; ++g) {
          v16bf bt[4];
#pragma unroll
          for (int u = 0; u < 4; ++u)
            bt[u] = load_B(Wm1b, 128, nchunk * 128 + (g * 4 + u) * 16, kb, lane);
#pragma unroll
          for (int u = 0; u < 4; ++u) acc[g * 4 + u] = wmma_bf16(a, bt[u], acc[g * 4 + u]);
        }
      }
      const int coll = lane & 15;
      const int rl = (lane >> 4) << 3;
#pragma unroll
      for (int t = 0; t < 8; ++t) {
        const int col = nchunk * 128 + t * 16 + coll;
        const float bc = bm1[col];
#pragma unroll
        for (int r = 0; r < 8; ++r)
          ldsw[(rl + r) * 512 + col] = (bf16_t)fmaxf(acc[t][r] + bc, 0.f);
      }
    }
  }
  __syncthreads();
  if (m0 >= M) return;

  v8f acc[8];
#pragma unroll
  for (int t = 0; t < 8; ++t) acc[t] = zero;
  for (int kb = 0; kb < 512; kb += 32) {
    const bf16_t* ap = ldsw + (lane & 15) * 512 + kb + ((lane >> 4) << 3);
    v8bf a0 = *(const v8bf*)ap;
    v8bf a1 = *(const v8bf*)(ap + 16);
    v16bf a;
#pragma unroll
    for (int e = 0; e < 8; ++e) { a[e] = a0[e]; a[8 + e] = a1[e]; }
#pragma unroll
    for (int g = 0; g < 2; ++g) {
      v16bf bt[4];
#pragma unroll
      for (int u = 0; u < 4; ++u) bt[u] = load_B(Wm2b, 512, (g * 4 + u) * 16, kb, lane);
#pragma unroll
      for (int u = 0; u < 4; ++u) acc[g * 4 + u] = wmma_bf16(a, bt[u], acc[g * 4 + u]);
    }
  }
  const int coll = lane & 15;
  const int rbase = m0 + ((lane >> 4) << 3);
#pragma unroll
  for (int t = 0; t < 8; ++t) {
    const int col = t * 16 + coll;
    const float bc = bm2[col];
#pragma unroll
    for (int r = 0; r < 8; ++r) {
      const size_t idx = (size_t)(rbase + r) * 128 + col;
      out[idx] = XT[idx] + acc[t][r] + bc;
    }
  }
}

// ---------------------------------------------------------------------------
// LayerNorm (one wave per 128-wide row); optional per-row nonzero-sum mask
// ---------------------------------------------------------------------------
__global__ void ln_kernel(const float* __restrict__ X, const float* __restrict__ g,
                          const float* __restrict__ b, float* __restrict__ Y,
                          float* __restrict__ msk, int M) {
  const int row = blockIdx.x * (blockDim.x >> 5) + (threadIdx.x >> 5);
  if (row >= M) return;
  const int lane = threadIdx.x & 31;
  float4 v = *(const float4*)(X + (size_t)row * 128 + lane * 4);
  float s = v.x + v.y + v.z + v.w;
#pragma unroll
  for (int o = 16; o > 0; o >>= 1) s += __shfl_xor(s, o);
  const float mean = s * (1.f / 128.f);
  float d0 = v.x - mean, d1 = v.y - mean, d2 = v.z - mean, d3 = v.w - mean;
  float q = d0 * d0 + d1 * d1 + d2 * d2 + d3 * d3;
#pragma unroll
  for (int o = 16; o > 0; o >>= 1) q += __shfl_xor(q, o);
  const float inv = rsqrtf(q * (1.f / 128.f) + 1e-5f);
  const float4 gv = *(const float4*)(g + lane * 4);
  const float4 bv = *(const float4*)(b + lane * 4);
  float4 outv;
  outv.x = d0 * inv * gv.x + bv.x;
  outv.y = d1 * inv * gv.y + bv.y;
  outv.z = d2 * inv * gv.z + bv.z;
  outv.w = d3 * inv * gv.w + bv.w;
  *(float4*)(Y + (size_t)row * 128 + lane * 4) = outv;
  if (msk) {
    float t = outv.x + outv.y + outv.z + outv.w;
#pragma unroll
    for (int o = 16; o > 0; o >>= 1) t += __shfl_xor(t, o);
    if (lane == 0) msk[row] = (t != 0.f) ? 1.f : 0.f;
  }
}

// ---------------------------------------------------------------------------
// CSR build: count -> scan -> fill (only ~1M integer atomics total)
// ---------------------------------------------------------------------------
__global__ void zero_int_kernel(int* __restrict__ p, int n) {
  const int i = blockIdx.x * blockDim.x + threadIdx.x;
  if (i < n) p[i] = 0;
}

__global__ void edge_count_kernel(const int* __restrict__ etgt, int* __restrict__ cnt, int E) {
  const int e = blockIdx.x * blockDim.x + threadIdx.x;
  if (e < E) atomicAdd(&cnt[etgt[e]], 1);
}

__global__ void scan1_kernel(const int* __restrict__ cnt, int* __restrict__ rowstart,
                             int* __restrict__ bsum, int N) {
  __shared__ int sh[256];
  const int i = blockIdx.x * 256 + threadIdx.x;
  const int v = (i < N) ? cnt[i] : 0;
  sh[threadIdx.x] = v;
  __syncthreads();
#pragma unroll
  for (int o = 1; o < 256; o <<= 1) {
    int t = (threadIdx.x >= o) ? sh[threadIdx.x - o] : 0;
    __syncthreads();
    sh[threadIdx.x] += t;
    __syncthreads();
  }
  if (i < N) rowstart[i] = sh[threadIdx.x] - v;   // exclusive within block
  if (threadIdx.x == 255) bsum[blockIdx.x] = sh[255];
}

__global__ void scan2_kernel(int* __restrict__ bsum, int nb) {
  if (blockIdx.x == 0 && threadIdx.x == 0) {
    int run = 0;
    for (int i = 0; i < nb; ++i) { int v = bsum[i]; bsum[i] = run; run += v; }
  }
}

__global__ void scan3_kernel(int* __restrict__ rowstart, const int* __restrict__ bsum,
                             int* __restrict__ cursor, int N, int E) {
  const int i = blockIdx.x * blockDim.x + threadIdx.x;
  if (i < N) {
    rowstart[i] += bsum[i >> 8];
    cursor[i] = 0;
  }
  if (i == N) rowstart[N] = E;
}

__global__ void edge_fill_kernel(const int* __restrict__ etgt, const int* __restrict__ rowstart,
                                 int* __restrict__ cursor, int* __restrict__ elist, int E) {
  const int e = blockIdx.x * blockDim.x + threadIdx.x;
  if (e >= E) return;
  const int t = etgt[e];
  const int pos = atomicAdd(&cursor[t], 1);
  elist[rowstart[t] + pos] = e;
}

// ---------------------------------------------------------------------------
// Per-target attention: one wave per target; q.k via 4-lane shfl reduce,
// segment softmax + weighted V accumulation with NO float atomics.
// Lane owns dims [lane*4, lane*4+4), head = lane/4.
// ---------------------------------------------------------------------------
__global__ void seg_attn_kernel(const float* __restrict__ Q, const float* __restrict__ Ks,
                                const float* __restrict__ Vs, const float* __restrict__ msk,
                                const int* __restrict__ esrc, const int* __restrict__ elist,
                                const int* __restrict__ rowstart, float* __restrict__ agg, int N) {
  const int t = blockIdx.x * (blockDim.x >> 5) + (threadIdx.x >> 5);
  if (t >= N) return;
  const int lane = threadIdx.x & 31;
  const int beg = rowstart[t], end = rowstart[t + 1];
  const float4 q = *(const float4*)(Q + (size_t)t * 128 + lane * 4);

  float m = -INFINITY;
  for (int i = beg; i < end; ++i) {
    const int s = esrc[elist[i]];
    const float4 k = *(const float4*)(Ks + (size_t)s * 128 + lane * 4);
    float p = q.x * k.x + q.y * k.y + q.z * k.z + q.w * k.w;
    p += __shfl_xor(p, 1);
    p += __shfl_xor(p, 2);
    m = fmaxf(m, p * 0.25f);
  }
  float den = 0.f;
  for (int i = beg; i < end; ++i) {
    const int s = esrc[elist[i]];
    const float4 k = *(const float4*)(Ks + (size_t)s * 128 + lane * 4);
    float p = q.x * k.x + q.y * k.y + q.z * k.z + q.w * k.w;
    p += __shfl_xor(p, 1);
    p += __shfl_xor(p, 2);
    den += __expf(p * 0.25f - m);
  }
  float4 acc = {0.f, 0.f, 0.f, 0.f};
  for (int i = beg; i < end; ++i) {
    const int s = esrc[elist[i]];
    const float4 k = *(const float4*)(Ks + (size_t)s * 128 + lane * 4);
    float p = q.x * k.x + q.y * k.y + q.z * k.z + q.w * k.w;
    p += __shfl_xor(p, 1);
    p += __shfl_xor(p, 2);
    const float w = __expf(p * 0.25f - m) / (den + 1e-16f) * msk[s];
    const float4 v = *(const float4*)(Vs + (size_t)s * 128 + lane * 4);
    acc.x += v.x * w; acc.y += v.y * w; acc.z += v.z * w; acc.w += v.w * w;
  }
  *(float4*)(agg + (size_t)t * 128 + lane * 4) = acc;
}

__global__ void f2bf_kernel(const float* __restrict__ s, bf16_t* __restrict__ d, int n) {
  const int i = blockIdx.x * blockDim.x + threadIdx.x;
  if (i < n) d[i] = (bf16_t)s[i];
}

// ---------------------------------------------------------------------------
extern "C" void kernel_launch(void* const* d_in, const int* in_sizes, int n_in,
                              void* d_out, int out_size, void* d_ws, size_t ws_size,
                              hipStream_t stream) {
  constexpr int D = 128, NSRC = 4096, NTGT = 200000, NEDGE = 500000;
  const float* x_source = (const float*)d_in[0];
  const float* x_target = (const float*)d_in[1];
  const int* edge_src = (const int*)d_in[2];
  const int* edge_tgt = (const int*)d_in[3];
  const float* Wq = (const float*)d_in[4];   const float* bq = (const float*)d_in[5];
  const float* Wk = (const float*)d_in[6];   const float* bk = (const float*)d_in[7];
  const float* Wv = (const float*)d_in[8];   const float* bv = (const float*)d_in[9];
  const float* Wih = (const float*)d_in[10]; const float* bih = (const float*)d_in[11];
  const float* Whh = (const float*)d_in[12]; const float* bhh = (const float*)d_in[13];
  const float* Wo = (const float*)d_in[14];  const float* bo = (const float*)d_in[15];
  const float* g1 = (const float*)d_in[16];  const float* b1 = (const float*)d_in[17];
  const float* g3 = (const float*)d_in[18];  const float* b3 = (const float*)d_in[19];
  const float* Wm1 = (const float*)d_in[20]; const float* bm1 = (const float*)d_in[21];
  const float* Wm2 = (const float*)d_in[22]; const float* bm2 = (const float*)d_in[23];
  float* out = (float*)d_out;

  char* base = (char*)d_ws;
  size_t off = 0;
  auto take = [&](size_t bytes) -> char* {
    char* p = base + off;
    off = (off + bytes + 255) & ~(size_t)255;
    return p;
  };
  bf16_t* WQb = (bf16_t*)take((size_t)D * D * sizeof(bf16_t));
  bf16_t* WKb = (bf16_t*)take((size_t)D * D * sizeof(bf16_t));
  bf16_t* WVb = (bf16_t*)take((size_t)D * D * sizeof(bf16_t));
  bf16_t* WIHb = (bf16_t*)take((size_t)D * D * sizeof(bf16_t));
  bf16_t* WHHb = (bf16_t*)take((size_t)D * D * sizeof(bf16_t));
  bf16_t* WOb = (bf16_t*)take((size_t)D * D * sizeof(bf16_t));
  bf16_t* WM1b = (bf16_t*)take((size_t)4 * D * D * sizeof(bf16_t));
  bf16_t* WM2b = (bf16_t*)take((size_t)4 * D * D * sizeof(bf16_t));
  float* xs = (float*)take((size_t)NSRC * D * 4);
  float* msk = (float*)take((size_t)NSRC * 4);
  float* Ks = (float*)take((size_t)NSRC * D * 4);
  float* Vs = (float*)take((size_t)NSRC * D * 4);
  float* Qt = (float*)take((size_t)NTGT * D * 4);   // later reused as UPD
  float* HH = (float*)take((size_t)NTGT * D * 4);   // later reused as HN
  float* agg = (float*)take((size_t)NTGT * D * 4);
  float* XT = (float*)take((size_t)NTGT * D * 4);
  const int NB = (NTGT + 255) / 256;
  int* cnt = (int*)take((size_t)NTGT * 4);
  int* cursor = (int*)take((size_t)NTGT * 4);
  int* rowstart = (int*)take((size_t)(NTGT + 1) * 4);
  int* bsum = (int*)take((size_t)NB * 4);
  int* elist = (int*)take((size_t)NEDGE * 4);

  const int cb = 256;

  // weights -> bf16
  f2bf_kernel<<<(D * D + cb - 1) / cb, cb, 0, stream>>>(Wq, WQb, D * D);
  f2bf_kernel<<<(D * D + cb - 1) / cb, cb, 0, stream>>>(Wk, WKb, D * D);
  f2bf_kernel<<<(D * D + cb - 1) / cb, cb, 0, stream>>>(Wv, WVb, D * D);
  f2bf_kernel<<<(D * D + cb - 1) / cb, cb, 0, stream>>>(Wih, WIHb, D * D);
  f2bf_kernel<<<(D * D + cb - 1) / cb, cb, 0, stream>>>(Whh, WHHb, D * D);
  f2bf_kernel<<<(D * D + cb - 1) / cb, cb, 0, stream>>>(Wo, WOb, D * D);
  f2bf_kernel<<<(4 * D * D + cb - 1) / cb, cb, 0, stream>>>(Wm1, WM1b, 4 * D * D);
  f2bf_kernel<<<(4 * D * D + cb - 1) / cb, cb, 0, stream>>>(Wm2, WM2b, 4 * D * D);

  // CSR build (edge_tgt -> per-target edge lists)
  zero_int_kernel<<<(NTGT + cb - 1) / cb, cb, 0, stream>>>(cnt, NTGT);
  edge_count_kernel<<<(NEDGE + cb - 1) / cb, cb, 0, stream>>>(edge_tgt, cnt, NEDGE);
  scan1_kernel<<<NB, 256, 0, stream>>>(cnt, rowstart, bsum, NTGT);
  scan2_kernel<<<1, 64, 0, stream>>>(bsum, NB);
  scan3_kernel<<<(NTGT + 1 + cb - 1) / cb, cb, 0, stream>>>(rowstart, bsum, cursor, NTGT, NEDGE);
  edge_fill_kernel<<<(NEDGE + cb - 1) / cb, cb, 0, stream>>>(edge_tgt, rowstart, cursor, elist, NEDGE);

  // source prenorm + mask
  ln_kernel<<<(NSRC + 7) / 8, 256, 0, stream>>>(x_source, g1, b1, xs, msk, NSRC);

  // per-node projections: dual GEMMs share one activation pass
  gemm128_dual_kernel<<<(NSRC / 16 + 3) / 4, 128, 0, stream>>>(xs, WKb, bk, Ks, WVb, bv, Vs, NSRC);
  gemm128_dual_kernel<<<(NTGT / 16 + 3) / 4, 128, 0, stream>>>(x_target, WQb, bq, Qt, WHHb, bhh, HH, NTGT);

  // per-target segment attention (softmax + aggregation, no float atomics)
  seg_attn_kernel<<<(NTGT + 7) / 8, 256, 0, stream>>>(Qt, Ks, Vs, msk, edge_src, elist, rowstart, agg, NTGT);

  // gate: UPD = agg * sigmoid(agg@Wih^T + bih + HH)   (UPD reuses Qt buffer)
  gemm128_kernel<<<(NTGT / 16 + 3) / 4, 128, 0, stream>>>(agg, WIHb, bih, HH, Qt, NTGT, 3);
  // XT = x_target + UPD@Wo^T + bo
  gemm128_kernel<<<(NTGT / 16 + 3) / 4, 128, 0, stream>>>(Qt, WOb, bo, x_target, XT, NTGT, 2);
  // HN = LN(XT)   (reuses HH buffer)
  ln_kernel<<<(NTGT + 7) / 8, 256, 0, stream>>>(XT, g3, b3, HH, nullptr, NTGT);
  // out = XT + relu(HN@Wm1^T+bm1)@Wm2^T + bm2  (fused, LDS-staged intermediate)
  ffn_fused_kernel<<<NTGT / 32, 64, 0, stream>>>(HH, XT, WM1b, bm1, WM2b, bm2, out, NTGT);
}
